// MultiHeadAttention_88699664597290
// MI455X (gfx1250) — matrix operand order repaired and checked by reading, not compiled, not run on hardware
//
#include <hip/hip_runtime.h>
#include <hip/hip_bf16.h>
#include <math.h>

// ---------------------------------------------------------------------------
// MHA forward for MI455X (gfx1250, wave32, WMMA bf16 16x16x32, f32 accum).
// B=2, S=2048, D=1024, H=16, DH=64.  d_out = [out fp32 4.19M][attn fp32 134M].
//
// Attention uses flash-style online softmax (2 WMMA passes over QK^T) so the
// only global traffic in the score strip is the mandatory fp32 attn write.
// The C-layout -> A-layout transpose of the probabilities goes through a
// per-wave 16x32 bf16 LDS strip (ds_store_b16 / ds_load_b128).
// ---------------------------------------------------------------------------

#define B_  2
#define S_  2048
#define D_  1024
#define H_  16
#define DH_ 64

typedef __attribute__((ext_vector_type(16))) __bf16        v16bf;
typedef __attribute__((ext_vector_type(8)))  float         v8f;
typedef __attribute__((ext_vector_type(4)))  float         v4f;
typedef __attribute__((ext_vector_type(4)))  unsigned int  v4u;

union FragBF {
  v16bf v;
  v4u   q[2];
  unsigned short u[16];
};

__device__ __forceinline__ unsigned short f2bf(float x) {
  unsigned int u = __float_as_uint(x);
  u += 0x7FFFu + ((u >> 16) & 1u);     // round-to-nearest-even
  return (unsigned short)(u >> 16);
}

__device__ __forceinline__ v8f zero8() {
  v8f z;
  for (int i = 0; i < 8; ++i) z[i] = 0.0f;
  return z;
}

// A-matrix 16x32 bf16 fragment (ISA 7.12.2): lanes 0-15 hold K = {k0..k0+7,
// k0+16..k0+23}; lanes 16-31 hold K = {k0+8..k0+15, k0+24..k0+31}.
__device__ __forceinline__ v16bf load_a16(const unsigned short* row, int k0, int hi) {
  FragBF f;
  f.q[0] = *reinterpret_cast<const v4u*>(row + k0 + hi * 8);
  f.q[1] = *reinterpret_cast<const v4u*>(row + k0 + 16 + hi * 8);
  return f.v;
}

// B-matrix 32x16 bf16 fragment: lane n holds column n; lanes 0-15 K={k0..k0+7,
// k0+16..k0+23}, lanes 16-31 K={k0+8..k0+15, k0+24..k0+31} (row-major NxK source = B^T).
__device__ __forceinline__ v16bf load_b16(const unsigned short* row, int k0, int hi) {
  FragBF f;
  f.q[0] = *reinterpret_cast<const v4u*>(row + k0 + hi * 8);
  f.q[1] = *reinterpret_cast<const v4u*>(row + k0 + 16 + hi * 8);
  return f.v;
}

__device__ __forceinline__ v8f wmma_bf16(v16bf a, v16bf b, v8f c) {
  return __builtin_amdgcn_wmma_f32_16x16x32_bf16(false, a, false, b,
                                                 (short)0, c, false, false);
}

// ---------------------------------------------------------------------------
// fp32 -> bf16 conversion
// ---------------------------------------------------------------------------
__global__ __launch_bounds__(256) void cvt_bf16_kernel(const float* __restrict__ in,
                                                       unsigned short* __restrict__ out,
                                                       int n) {
  int i = blockIdx.x * blockDim.x + threadIdx.x;
  if (i < n) out[i] = f2bf(in[i]);
}

// ---------------------------------------------------------------------------
// y = x @ W^T + b   (M=4096, N=1024, K=1024), bf16 WMMA, f32 accum.
// mode 0: store bf16 head-major [B,H,S,DH]   (Q, K projections)
// mode 1: store bf16 transposed [B,H,DH,S]   (V projection)
// mode 2: store fp32 [B*S, D]                (output projection)
// One wave -> 16 rows x 64 cols (4 N-tiles sharing the A fragment).
// ---------------------------------------------------------------------------
__global__ __launch_bounds__(256) void proj_kernel(const unsigned short* __restrict__ X,
                                                   const unsigned short* __restrict__ W,
                                                   const float* __restrict__ bias,
                                                   unsigned short* __restrict__ outB,
                                                   float* __restrict__ outF,
                                                   int mode) {
  const int lane  = threadIdx.x & 31;
  const int hi    = lane >> 4;
  const int lo    = lane & 15;
  const int wid   = blockIdx.x * 8 + (threadIdx.x >> 5);   // 0..4095
  const int mtile = wid >> 4;                              // 0..255
  const int ng    = wid & 15;                              // 0..15 (64-col group)

  const unsigned short* arow = X + (size_t)(mtile * 16 + lo) * D_;

  v8f acc[4];
  for (int t = 0; t < 4; ++t) acc[t] = zero8();

  for (int kc = 0; kc < D_ / 32; ++kc) {
    const int k0 = kc * 32;
    v16bf a = load_a16(arow, k0, hi);
    for (int t = 0; t < 4; ++t) {
      const unsigned short* brow = W + (size_t)(ng * 64 + t * 16 + lo) * D_;
      acc[t] = wmma_bf16(a, load_b16(brow, k0, hi), acc[t]);
    }
  }

  for (int t = 0; t < 4; ++t) {
    const int col = ng * 64 + t * 16 + lo;     // C layout: lane -> column
    const float bv = bias[col];
    const int h = col >> 6, d = col & 63;
    for (int i = 0; i < 8; ++i) {
      const int m = i + hi * 8;                // C layout: vgpr(+8*hi) -> row
      const int r = mtile * 16 + m;
      const float val = acc[t][i] + bv;
      if (mode == 0) {
        const int b = r >> 11, s = r & (S_ - 1);
        outB[(((size_t)b * H_ + h) * S_ + s) * DH_ + d] = f2bf(val);
      } else if (mode == 1) {
        const int b = r >> 11, s = r & (S_ - 1);
        outB[(((size_t)b * H_ + h) * DH_ + d) * S_ + s] = f2bf(val);
      } else {
        outF[(size_t)r * D_ + col] = val;
      }
    }
  }
}

// ---------------------------------------------------------------------------
// Attention: one wave per (b,h, 16-query tile).  Online softmax, 2 WMMA passes.
// ---------------------------------------------------------------------------
__global__ __launch_bounds__(256) void mha_attn_kernel(
    const unsigned short* __restrict__ qh,   // bf16 [B,H,S,DH]
    const unsigned short* __restrict__ kh,   // bf16 [B,H,S,DH]
    const unsigned short* __restrict__ vT,   // bf16 [B,H,DH,S]
    float* __restrict__ attn,                // f32  [B,H,S,S]
    unsigned short* __restrict__ ctxb) {     // bf16 [B,S,D]
  __shared__ unsigned short pbuf[8][16 * 32];  // per-wave 16x32 bf16 strip

  const int lane  = threadIdx.x & 31;
  const int hi    = lane >> 4;
  const int lo    = lane & 15;
  const int wslot = threadIdx.x >> 5;
  const int wid   = blockIdx.x * 8 + wslot;  // 0..4095
  const int bh    = wid >> 7;                // 0..31
  const int qt    = wid & 127;
  const int qrow0 = qt << 4;
  const int kmax  = qrow0 + 16;              // causal: cols [0, kmax) reachable

  const unsigned short* qbase = qh + (size_t)bh * S_ * DH_;
  const unsigned short* kbase = kh + (size_t)bh * S_ * DH_;
  const unsigned short* vbase = vT + (size_t)bh * DH_ * S_;
  float* arow = attn + ((size_t)bh * S_ + qrow0) * S_;

  const float NEG = -1.0e30f;

  // Q fragments are invariant across the whole kernel.
  const unsigned short* qrow = qbase + (size_t)(qrow0 + lo) * DH_;
  const v16bf a0 = load_a16(qrow, 0, hi);
  const v16bf a1 = load_a16(qrow, 32, hi);

  // ---- pass 1: online max / sum per (row, lane-column-slot), C layout ----
  float m[8], l[8];
  for (int i = 0; i < 8; ++i) { m[i] = NEG; l[i] = 0.0f; }

  for (int kt = 0; kt < (kmax >> 4); ++kt) {
    const unsigned short* krow = kbase + (size_t)(kt * 16 + lo) * DH_;
    v8f acc = zero8();
    acc = wmma_bf16(a0, load_b16(krow, 0, hi), acc);
    acc = wmma_bf16(a1, load_b16(krow, 32, hi), acc);
    const int col = kt * 16 + lo;
    for (int i = 0; i < 8; ++i) {
      const int mr = i + hi * 8;
      if (col <= qrow0 + mr) {               // unmasked only (skip -> no 1-add)
        const float s  = acc[i] * 0.125f;
        const float mn = fmaxf(m[i], s);
        l[i] = l[i] * __expf(m[i] - mn) + __expf(s - mn);
        m[i] = mn;
      }
    }
  }
  // Cross-lane (16-lane half) reduction: row max M_i and row sum L_i.
  float rinv[8];
  for (int i = 0; i < 8; ++i) {
    float M = m[i];
    for (int d = 1; d < 16; d <<= 1) M = fmaxf(M, __shfl_xor(M, d, 32));
    float ls = l[i] * __expf(m[i] - M);      // rescale partial sum to final max
    for (int d = 1; d < 16; d <<= 1) ls += __shfl_xor(ls, d, 32);
    m[i]    = M;
    rinv[i] = 1.0f / ls;
  }

  // ---- zero-fill the masked tail of the attn rows (output is poisoned) ----
  {
    float* prow = arow + (size_t)lo * S_;    // row lo; two lanes per row
    v4f z;
    for (int j = 0; j < 4; ++j) z[j] = 0.0f;
    for (int col = kmax + hi * 16; col < S_; col += 32) {
      *reinterpret_cast<v4f*>(prow + col)      = z;
      *reinterpret_cast<v4f*>(prow + col + 4)  = z;
      *reinterpret_cast<v4f*>(prow + col + 8)  = z;
      *reinterpret_cast<v4f*>(prow + col + 12) = z;
    }
  }

  // ---- pass 2: recompute scores, emit normalized attn, ctx = attn @ V ----
  unsigned short* pl = &pbuf[wslot][0];
  v8f cacc[4];
  for (int t = 0; t < 4; ++t) cacc[t] = zero8();

  const int nch = (kmax + 31) >> 5;
  for (int c = 0; c < nch; ++c) {
    const int base = c * 32;
    for (int t2 = 0; t2 < 2; ++t2) {
      const int col0 = base + t2 * 16;
      if (col0 < kmax) {
        const unsigned short* krow = kbase + (size_t)(col0 + lo) * DH_;
        v8f acc = zero8();
        acc = wmma_bf16(a0, load_b16(krow, 0, hi), acc);
        acc = wmma_bf16(a1, load_b16(krow, 32, hi), acc);
        const int col = col0 + lo;
        for (int i = 0; i < 8; ++i) {
          const int mr = i + hi * 8;
          const float pn = (col <= qrow0 + mr)
                               ? __expf(acc[i] * 0.125f - m[i]) * rinv[i]
                               : 0.0f;
          arow[(size_t)mr * S_ + col]  = pn;           // fp32 attn output
          pl[mr * 32 + t2 * 16 + lo]   = f2bf(pn);     // bf16 strip in LDS
        }
      } else {
        for (int i = 0; i < 8; ++i)                    // keep A-frag clean
          pl[(i + hi * 8) * 32 + t2 * 16 + lo] = 0;
      }
    }
    // A fragment from LDS (transpose: lane -> probability row lo).
    FragBF a;
    a.q[0] = *reinterpret_cast<const v4u*>(pl + lo * 32 + hi * 8);
    a.q[1] = *reinterpret_cast<const v4u*>(pl + lo * 32 + 16 + hi * 8);
    for (int t = 0; t < 4; ++t) {
      const unsigned short* vrow = vbase + (size_t)(t * 16 + lo) * S_;
      cacc[t] = wmma_bf16(a.v, load_b16(vrow, base, hi), cacc[t]);
    }
  }

  const int b = bh >> 4, h = bh & 15;
  for (int t = 0; t < 4; ++t)
    for (int i = 0; i < 8; ++i) {
      const int mr = i + hi * 8;
      ctxb[((size_t)b * S_ + qrow0 + mr) * D_ + h * 64 + t * 16 + lo] =
          f2bf(cacc[t][i]);
    }
}

// ---------------------------------------------------------------------------
extern "C" void kernel_launch(void* const* d_in, const int* in_sizes, int n_in,
                              void* d_out, int out_size, void* d_ws, size_t ws_size,
                              hipStream_t stream) {
  (void)in_sizes; (void)n_in; (void)out_size; (void)ws_size;

  const float* q  = (const float*)d_in[0];
  const float* k  = (const float*)d_in[1];
  const float* v  = (const float*)d_in[2];
  // d_in[3] = mask (causal, structure known -> unused)
  const float* wq = (const float*)d_in[4];
  const float* bq = (const float*)d_in[5];
  const float* wk = (const float*)d_in[6];
  const float* bk = (const float*)d_in[7];
  const float* wv = (const float*)d_in[8];
  const float* bv = (const float*)d_in[9];
  const float* wo = (const float*)d_in[10];
  const float* bo = (const float*)d_in[11];

  float* out  = (float*)d_out;                               // [B,S,D]
  float* attn = out + (size_t)B_ * S_ * D_;                  // [B,H,S,S]

  const size_t NX = (size_t)B_ * S_ * D_;   // 4,194,304 elements
  const size_t NW = (size_t)D_ * D_;        // 1,048,576 elements

  char* ws = (char*)d_ws;
  unsigned short* xq   = (unsigned short*)ws; ws += NX * 2;
  unsigned short* xk   = (unsigned short*)ws; ws += NX * 2;
  unsigned short* xv   = (unsigned short*)ws; ws += NX * 2;
  unsigned short* wqb  = (unsigned short*)ws; ws += NW * 2;
  unsigned short* wkb  = (unsigned short*)ws; ws += NW * 2;
  unsigned short* wvb  = (unsigned short*)ws; ws += NW * 2;
  unsigned short* wob  = (unsigned short*)ws; ws += NW * 2;
  unsigned short* qhb  = (unsigned short*)ws; ws += NX * 2;
  unsigned short* khb  = (unsigned short*)ws; ws += NX * 2;
  unsigned short* vTb  = (unsigned short*)ws; ws += NX * 2;
  unsigned short* ctxb = (unsigned short*)ws; ws += NX * 2;   // ~64 MB total

  const int CT = 256;
  const int GX = (int)((NX + CT - 1) / CT);
  const int GW = (int)((NW + CT - 1) / CT);
  cvt_bf16_kernel<<<GX, CT, 0, stream>>>(q,  xq,  (int)NX);
  cvt_bf16_kernel<<<GX, CT, 0, stream>>>(k,  xk,  (int)NX);
  cvt_bf16_kernel<<<GX, CT, 0, stream>>>(v,  xv,  (int)NX);
  cvt_bf16_kernel<<<GW, CT, 0, stream>>>(wq, wqb, (int)NW);
  cvt_bf16_kernel<<<GW, CT, 0, stream>>>(wk, wkb, (int)NW);
  cvt_bf16_kernel<<<GW, CT, 0, stream>>>(wv, wvb, (int)NW);
  cvt_bf16_kernel<<<GW, CT, 0, stream>>>(wo, wob, (int)NW);

  // 4096 waves each: (M/16)*(N/64) tiles; 8 waves per 256-thread block.
  proj_kernel<<<512, 256, 0, stream>>>(xq, wqb, bq, qhb, nullptr, 0);
  proj_kernel<<<512, 256, 0, stream>>>(xk, wkb, bk, khb, nullptr, 0);
  proj_kernel<<<512, 256, 0, stream>>>(xv, wvb, bv, vTb, nullptr, 1);

  // B*H*(S/16) = 4096 waves
  mha_attn_kernel<<<512, 256, 0, stream>>>(qhb, khb, vTb, attn, ctxb);

  proj_kernel<<<512, 256, 0, stream>>>(ctxb, wob, bo, nullptr, out, 2);
}
